// IterRCNNHead_85916525789874
// MI455X (gfx1250) — matrix operand order repaired and checked by reading, not compiled, not run on hardware
//
#include <hip/hip_runtime.h>

// ---------------- problem constants ----------------
#define Nn  8
#define NR  300
#define Dd  256
#define Hh  8
#define Ss  49
#define DDd 64
#define FFf 2048
#define NCc 80
#define HD  32
#define NT  (Nn*NR)          // 2400
#define PAR (2*Dd*DDd)       // 32768
#define SD  (Ss*Dd)          // 12544

typedef __attribute__((ext_vector_type(16))) __bf16 v16bf;
typedef __attribute__((ext_vector_type(8)))  float  v8f;
typedef __attribute__((ext_vector_type(4)))  unsigned int v4u;
typedef __attribute__((ext_vector_type(8)))  int v8i;
typedef __attribute__((ext_vector_type(4)))  int v4i;

// Tensor Data Mover availability (arity differs across toolchains)
#if defined(__has_builtin)
#if __has_builtin(__builtin_amdgcn_tensor_load_to_lds) && __has_builtin(__builtin_amdgcn_s_wait_tensorcnt)
#define USE_TDM 1
#endif
#endif

// ---------------- small helpers ----------------
__device__ inline unsigned short f2bf(float f) {
    union { float f; unsigned int u; } v; v.f = f;
    unsigned int u = v.u;
    unsigned int r = u + 0x7FFFu + ((u >> 16) & 1u);   // round-to-nearest-even
    return (unsigned short)(r >> 16);
}
__device__ inline __bf16 bfrom(unsigned short u) {
    return __builtin_bit_cast(__bf16, u);
}
__device__ inline float wsum(float v) {
#pragma unroll
    for (int m = 16; m > 0; m >>= 1) v += __shfl_xor(v, m, 32);
    return v;
}
__device__ inline float wmaxr(float v) {
#pragma unroll
    for (int m = 16; m > 0; m >>= 1) v = fmaxf(v, __shfl_xor(v, m, 32));
    return v;
}
__device__ inline v8f wmma_bf16(v16bf a, v16bf b, v8f c) {
    return __builtin_amdgcn_wmma_f32_16x16x32_bf16(false, a, false, b, (short)0, c, false, false);
}
// A fragment from LDS tile (row-major, stride elems). element (m,k):
//   m = lane&15 ; k = 16*(lane>>4) + j
__device__ inline v16bf frag_a_lds(const unsigned short* base, int stride, int lane) {
    int m  = lane & 15;
    int kb = (lane >> 4) << 4;
    v16bf a;
#pragma unroll
    for (int j = 0; j < 16; ++j) {
        int k = kb + j;
        a[j] = bfrom(base[m * stride + k]);
    }
    return a;
}
// B fragment from LDS tile of W rows. element (n,k): n = lane&15 ; k = 16*(lane>>4)+j
__device__ inline v16bf frag_b_lds(const unsigned short* base, int stride, int lane) {
    int n  = lane & 15;
    int kb = (lane >> 4) << 4;
    v16bf b;
#pragma unroll
    for (int j = 0; j < 16; ++j) b[j] = bfrom(base[n * stride + kb + j]);
    return b;
}

#ifdef USE_TDM
// Issue one TDM 2-D tile load: tile_dim0 = 32 elems (bf16, contiguous K),
// tile_dim1 = 64 rows, row stride = ld elems; rows beyond rows_total zero-fill.
__device__ inline void tdm_load_tile(unsigned ldsoff, const void* gaddr,
                                     unsigned k_total, unsigned rows_total,
                                     unsigned long long ld) {
    unsigned long long ga = (unsigned long long)(uintptr_t)gaddr;
    v4u g0;
    g0[0] = 1u;                                                 // count=1, user mode
    g0[1] = ldsoff;                                             // lds_addr
    g0[2] = (unsigned)ga;                                       // global_addr[31:0]
    g0[3] = (unsigned)((ga >> 32) & 0x01FFFFFFu) | 0x80000000u; // addr[56:32] | type=2
    v8i g1;
    g1[0] = 0x00010000;                                         // data_size = 2 bytes
    g1[1] = (int)((k_total & 0xFFFFu) << 16);                   // tensor_dim0 lo
    g1[2] = (int)((k_total >> 16) | ((rows_total & 0xFFFFu) << 16)); // td0 hi | td1 lo
    g1[3] = (int)(((rows_total >> 16) & 0xFFFFu) | (32u << 16));     // td1 hi | tile_dim0=32
    g1[4] = 64;                                                 // tile_dim1=64, tile_dim2=0
    g1[5] = (int)(ld & 0xFFFFFFFFull);                          // dim0 stride lo
    g1[6] = (int)((ld >> 32) & 0xFFFFull);                      // dim0 stride hi
    g1[7] = 0;
    v4i gz = {0, 0, 0, 0};
#if __clang_major__ >= 23
    v8i gz8 = {0, 0, 0, 0, 0, 0, 0, 0};
    __builtin_amdgcn_tensor_load_to_lds(g0, g1, gz, gz, gz8, 0);
#else
    __builtin_amdgcn_tensor_load_to_lds(g0, g1, gz, gz, 0);
#endif
}
#endif

// ---------------- fp32 -> bf16 conversion ----------------
__global__ void f32_to_bf16_kernel(const float* __restrict__ src,
                                   unsigned short* __restrict__ dst, long long n) {
    long long i = (long long)blockIdx.x * blockDim.x + threadIdx.x;
    long long st = (long long)gridDim.x * blockDim.x;
    for (; i < n; i += st) dst[i] = f2bf(src[i]);
}

// qk = bf16(pro+query); pf = bf16(pro)
__global__ void qkprep_kernel(const float* __restrict__ pro, const float* __restrict__ query,
                              unsigned short* __restrict__ qkb, unsigned short* __restrict__ pfb,
                              long long n) {
    long long i = (long long)blockIdx.x * blockDim.x + threadIdx.x;
    long long st = (long long)gridDim.x * blockDim.x;
    for (; i < n; i += st) {
        float p = pro[i];
        qkb[i] = f2bf(p + query[i]);
        pfb[i] = f2bf(p);
    }
}

// ---------------- IoU-relation mask -> additive bias ----------------
__global__ void mask_bias_kernel(const float* __restrict__ bboxes,
                                 const float* __restrict__ cur,
                                 float* __restrict__ mb) {
    int q = blockIdx.x % NR;
    int b = blockIdx.x / NR;
    const float* B = bboxes + (size_t)b * NR * 4;
    float qx0 = B[q*4+0], qy0 = B[q*4+1], qx1 = B[q*4+2], qy1 = B[q*4+3];
    float aq = (qx1 - qx0) * (qy1 - qy0);
    float cq = cur[b * NR + q];
    for (int k = threadIdx.x; k < NR; k += blockDim.x) {
        float kx0 = B[k*4+0], ky0 = B[k*4+1], kx1 = B[k*4+2], ky1 = B[k*4+3];
        float ak = (kx1 - kx0) * (ky1 - ky0);
        float iw = fmaxf(fminf(qx1, kx1) - fmaxf(qx0, kx0), 0.f);
        float ih = fmaxf(fminf(qy1, ky1) - fmaxf(qy0, ky0), 0.f);
        float inter = iw * ih;
        float uni = aq + ak - inter;
        float iou = inter / fmaxf(uni, 1e-9f);
        float om  = (iou < 0.5f) ? 1.f : 0.f;
        float pr  = cq * cur[b * NR + k];
        float eye = (q == k) ? 1.f : 0.f;
        float am  = om * pr + eye * (1.f - cq);
        mb[((size_t)b * NR + q) * NR + k] = (am > 0.f) ? -1e9f : 0.f;
    }
}

// ---------------- generic tiled WMMA GEMM ----------------
// C[M,N] = act(A[M,K]bf16 @ W[N,K]bf16^T + bias); TDM double-buffered staging.
__launch_bounds__(256)
__global__ void gemm_bf16_kernel(const unsigned short* __restrict__ A, int lda,
                                 const unsigned short* __restrict__ W, int ldw,
                                 const float* __restrict__ bias,
                                 float* __restrict__ outF, unsigned short* __restrict__ outB,
                                 int ldc, int M, int N, int K, int relu) {
    int bm = blockIdx.x * 64, bn = blockIdx.y * 64;
    int tid = threadIdx.x, wave = tid >> 5, lane = tid & 31;
    int mt = wave >> 1;
    int nh = (wave & 1) * 32;
    v8f c0 = {}, c1 = {};
#ifdef USE_TDM
    __shared__ unsigned short As[2][64][32];
    __shared__ unsigned short Bs[2][64][32];
    int nk = K >> 5;
    if (wave == 0) {
        tdm_load_tile((unsigned)(uintptr_t)&As[0][0][0], A + (size_t)bm * lda,
                      (unsigned)K, (unsigned)(M - bm), (unsigned long long)lda);
        tdm_load_tile((unsigned)(uintptr_t)&Bs[0][0][0], W + (size_t)bn * ldw,
                      (unsigned)K, (unsigned)(N - bn), (unsigned long long)ldw);
    }
    for (int s = 0; s < nk; ++s) {
        int cur = s & 1;
        if (wave == 0) {
            if (s + 1 < nk) {
                int nxt = cur ^ 1;
                tdm_load_tile((unsigned)(uintptr_t)&As[nxt][0][0],
                              A + (size_t)bm * lda + (s + 1) * 32,
                              (unsigned)K, (unsigned)(M - bm), (unsigned long long)lda);
                tdm_load_tile((unsigned)(uintptr_t)&Bs[nxt][0][0],
                              W + (size_t)bn * ldw + (s + 1) * 32,
                              (unsigned)K, (unsigned)(N - bn), (unsigned long long)ldw);
                __builtin_amdgcn_s_wait_tensorcnt(2);   // current tile pair complete
            } else {
                __builtin_amdgcn_s_wait_tensorcnt(0);
            }
        }
        __syncthreads();
        v16bf a  = frag_a_lds(&As[cur][mt * 16][0], 32, lane);
        v16bf b0 = frag_b_lds(&Bs[cur][nh][0],      32, lane);
        v16bf b1 = frag_b_lds(&Bs[cur][nh + 16][0], 32, lane);
        c0 = wmma_bf16(a, b0, c0);
        c1 = wmma_bf16(a, b1, c1);
        __syncthreads();
    }
#else
    __shared__ unsigned short As[64][32];
    __shared__ unsigned short Bs[64][32];
    int lr = tid >> 2;            // tile row loaded by this thread
    int lc = (tid & 3) << 3;      // first col (8 ushort = 16B)
    for (int kk = 0; kk < K; kk += 32) {
        {
            int gr = bm + lr;
            uint4 v = {0, 0, 0, 0};
            if (gr < M) v = *(const uint4*)(A + (size_t)gr * lda + kk + lc);
            *(uint4*)(&As[lr][lc]) = v;
            if (gr < M && kk + 32 < K)
                __builtin_prefetch(A + (size_t)gr * lda + kk + 32 + lc, 0, 1);
        }
        {
            int gn = bn + lr;
            uint4 v = {0, 0, 0, 0};
            if (gn < N) v = *(const uint4*)(W + (size_t)gn * ldw + kk + lc);
            *(uint4*)(&Bs[lr][lc]) = v;
            if (gn < N && kk + 32 < K)
                __builtin_prefetch(W + (size_t)gn * ldw + kk + 32 + lc, 0, 1);
        }
        __syncthreads();
        v16bf a  = frag_a_lds(&As[mt * 16][0], 32, lane);
        v16bf b0 = frag_b_lds(&Bs[nh][0],      32, lane);
        v16bf b1 = frag_b_lds(&Bs[nh + 16][0], 32, lane);
        c0 = wmma_bf16(a, b0, c0);
        c1 = wmma_bf16(a, b1, c1);
        __syncthreads();
    }
#endif
    int col0 = bn + nh + (lane & 15);
    int col1 = col0 + 16;
    int rb = bm + mt * 16 + ((lane >> 4) << 3);
#pragma unroll
    for (int j = 0; j < 8; ++j) {
        int gr = rb + j;
        if (gr >= M) continue;
        if (col0 < N) {
            float v = c0[j] + (bias ? bias[col0] : 0.f);
            if (relu) v = fmaxf(v, 0.f);
            if (outF) outF[(size_t)gr * ldc + col0] = v;
            if (outB) outB[(size_t)gr * ldc + col0] = f2bf(v);
        }
        if (col1 < N) {
            float v = c1[j] + (bias ? bias[col1] : 0.f);
            if (relu) v = fmaxf(v, 0.f);
            if (outF) outF[(size_t)gr * ldc + col1] = v;
            if (outB) outB[(size_t)gr * ldc + col1] = f2bf(v);
        }
    }
}

// ---------------- masked multi-head attention ----------------
// one wave per (image b, head h, 16-query tile); 19 key tiles padded to 320
__launch_bounds__(32)
__global__ void attn_kernel(const unsigned short* __restrict__ qkv,   // [NT,768] bf16
                            const float* __restrict__ mb,             // [N,NR,NR] bias
                            unsigned short* __restrict__ ctx) {       // [NT,256] bf16
    __shared__ float          sc[16][320];
    __shared__ unsigned short ps[16][320];
    int bid = blockIdx.x;
    int qt = bid % 19; int t = bid / 19;
    int h = t % Hh;    int b = t / Hh;
    int lane = threadIdx.x;
    int q0 = qt * 16;
    const float scale = 0.17677669529663687f;   // 1/sqrt(32)
    if (lane < 16) {
#pragma unroll
        for (int k = 304; k < 320; ++k) sc[lane][k] = -1e9f;
    }
    // --- scores ---
    for (int kt = 0; kt < 19; ++kt) {
        int qr = q0 + (lane & 15);
        int kb = (lane >> 4) << 4;
        v16bf a;
#pragma unroll
        for (int j = 0; j < 16; ++j) {
            int k = kb + j;
            unsigned short u = (qr < NR) ? qkv[((size_t)(b * NR + qr)) * 768 + h * HD + k] : 0;
            a[j] = bfrom(u);
        }
        int key = kt * 16 + (lane & 15);
        int kbb = (lane >> 4) << 4;
        v16bf bb;
#pragma unroll
        for (int j = 0; j < 16; ++j) {
            unsigned short u = (key < NR)
                ? qkv[((size_t)(b * NR + key)) * 768 + Dd + h * HD + kbb + j] : 0;
            bb[j] = bfrom(u);
        }
        v8f c = {};
        c = wmma_bf16(a, bb, c);
#pragma unroll
        for (int j = 0; j < 8; ++j) {
            int r = j + ((lane >> 4) << 3);
            int qq = q0 + r;
            float bias = -1e9f;
            if (qq < NR && key < NR) bias = mb[((size_t)b * NR + qq) * NR + key];
            sc[r][kt * 16 + (lane & 15)] = c[j] * scale + bias;
        }
    }
    __syncthreads();
    // --- softmax over 320 (pads = -1e9) ---
    for (int r = 0; r < 16; ++r) {
        float m = -1e30f;
        for (int k = lane; k < 320; k += 32) m = fmaxf(m, sc[r][k]);
        m = wmaxr(m);
        float s = 0.f;
        for (int k = lane; k < 320; k += 32) s += expf(sc[r][k] - m);
        s = wsum(s);
        float inv = 1.f / s;
        for (int k = lane; k < 320; k += 32) ps[r][k] = f2bf(expf(sc[r][k] - m) * inv);
    }
    __syncthreads();
    // --- ctx = P @ V ---
    v8f c0 = {}, c1 = {};
    for (int kt2 = 0; kt2 < 10; ++kt2) {
        int kbase = kt2 * 32;
        int row = lane & 15;
        int kb = (lane >> 4) << 4;
        v16bf a;
#pragma unroll
        for (int j = 0; j < 16; ++j) {
            int k = kbase + kb + j;
            a[j] = bfrom(ps[row][k]);
        }
        int kbb = (lane >> 4) << 4;
        v16bf b0, b1;
#pragma unroll
        for (int j = 0; j < 16; ++j) {
            int key = kbase + kbb + j;
            unsigned short u0 = 0, u1 = 0;
            if (key < NR) {
                size_t base = ((size_t)(b * NR + key)) * 768 + 2 * Dd + h * HD;
                u0 = qkv[base + (lane & 15)];
                u1 = qkv[base + 16 + (lane & 15)];
            }
            b0[j] = bfrom(u0);
            b1[j] = bfrom(u1);
        }
        c0 = wmma_bf16(a, b0, c0);
        c1 = wmma_bf16(a, b1, c1);
    }
#pragma unroll
    for (int j = 0; j < 8; ++j) {
        int q = q0 + j + ((lane >> 4) << 3);
        if (q >= NR) continue;
        size_t base = ((size_t)(b * NR + q)) * Dd + h * HD;
        ctx[base + (lane & 15)]      = f2bf(c0[j]);
        ctx[base + 16 + (lane & 15)] = f2bf(c1[j]);
    }
}

// ---------------- row LayerNorm (cols == 256) ----------------
__launch_bounds__(256)
__global__ void ln_rows_kernel(const float* __restrict__ x, const float* __restrict__ resid,
                               const float* __restrict__ rowmask,
                               float* __restrict__ outF, unsigned short* __restrict__ outB,
                               int rows, int relu) {
    int wave = threadIdx.x >> 5, lane = threadIdx.x & 31;
    int row = blockIdx.x * 8 + wave;
    if (row >= rows) return;
    const float* xr = x + (size_t)row * Dd;
    float v[8];
    float loc = 0.f;
#pragma unroll
    for (int i = 0; i < 8; ++i) {
        float t = xr[lane + 32 * i];
        if (resid) t += resid[(size_t)row * Dd + lane + 32 * i];
        v[i] = t; loc += t;
    }
    float mean = wsum(loc) * (1.f / Dd);
    float vv = 0.f;
#pragma unroll
    for (int i = 0; i < 8; ++i) { float d = v[i] - mean; vv += d * d; }
    float inv = rsqrtf(wsum(vv) * (1.f / Dd) + 1e-5f);
    float mk = rowmask ? rowmask[row] : 1.f;
#pragma unroll
    for (int i = 0; i < 8; ++i) {
        float y = (v[i] - mean) * inv;
        if (relu) y = fmaxf(y, 0.f);
        y *= mk;
        if (outF) outF[(size_t)row * Dd + lane + 32 * i] = y;
        if (outB) outB[(size_t)row * Dd + lane + 32 * i] = f2bf(y);
    }
}

// z = ln(resid + relu(ln(g)))  (norm2 stage), cols == 256
__launch_bounds__(256)
__global__ void ln_relu_ln_kernel(const float* __restrict__ g, const float* __restrict__ resid,
                                  float* __restrict__ outF, unsigned short* __restrict__ outB,
                                  int rows) {
    int wave = threadIdx.x >> 5, lane = threadIdx.x & 31;
    int row = blockIdx.x * 8 + wave;
    if (row >= rows) return;
    float v[8];
    float loc = 0.f;
#pragma unroll
    for (int i = 0; i < 8; ++i) { v[i] = g[(size_t)row * Dd + lane + 32 * i]; loc += v[i]; }
    float mean = wsum(loc) * (1.f / Dd);
    float vv = 0.f;
#pragma unroll
    for (int i = 0; i < 8; ++i) { float d = v[i] - mean; vv += d * d; }
    float inv = rsqrtf(wsum(vv) * (1.f / Dd) + 1e-5f);
    loc = 0.f;
#pragma unroll
    for (int i = 0; i < 8; ++i) {
        float y = fmaxf((v[i] - mean) * inv, 0.f) + resid[(size_t)row * Dd + lane + 32 * i];
        v[i] = y; loc += y;
    }
    mean = wsum(loc) * (1.f / Dd);
    vv = 0.f;
#pragma unroll
    for (int i = 0; i < 8; ++i) { float d = v[i] - mean; vv += d * d; }
    inv = rsqrtf(wsum(vv) * (1.f / Dd) + 1e-5f);
#pragma unroll
    for (int i = 0; i < 8; ++i) {
        float y = (v[i] - mean) * inv;
        if (outF) outF[(size_t)row * Dd + lane + 32 * i] = y;
        if (outB) outB[(size_t)row * Dd + lane + 32 * i] = f2bf(y);
    }
}

// ---------------- fused DynamicConv (both einsums + both LN/ReLU) ----------------
__launch_bounds__(256)
__global__ void dyn_interact_kernel(const unsigned short* __restrict__ roi_bf, // [S,NT,D]
                                    const unsigned short* __restrict__ params, // [NT,32768]
                                    unsigned short* __restrict__ f2out) {      // [NT,12544]
    extern __shared__ char smem[];
    float (*t1)[64]           = (float (*)[64])smem;                       // 16 KB
    unsigned short (*t1b)[64] = (unsigned short (*)[64])(smem + 16384);    //  8 KB
    float (*t2)[256]          = (float (*)[256])(smem + 16384 + 8192);     // 64 KB
    int n = blockIdx.x;
    int tid = threadIdx.x, wave = tid >> 5, lane = tid & 31;
    const unsigned short* pp = params + (size_t)n * PAR;
    // ---- phase 1: T1[49(pad64),64] = feats[49,256] @ p1[256,64] ----
    {
        int mt = wave >> 1;
        int nh = (wave & 1) * 32;
        v8f c0 = {}, c1 = {};
        int s = mt * 16 + (lane & 15);
        for (int kk = 0; kk < Dd; kk += 32) {
            if (kk + 32 < Dd)
                __builtin_prefetch(pp + (kk + 32) * DDd + lane * 8, 0, 1);
            int kb = (lane >> 4) << 4;
            v16bf a;
#pragma unroll
            for (int j = 0; j < 16; ++j) {
                int k = kk + kb + j;
                unsigned short u = (s < Ss) ? roi_bf[((size_t)s * NT + n) * Dd + k] : 0;
                a[j] = bfrom(u);
            }
            int kbb = (lane >> 4) << 4;
            v16bf b0, b1;
#pragma unroll
            for (int j = 0; j < 16; ++j) {
                int k = kk + kbb + j;
                b0[j] = bfrom(pp[k * DDd + nh + (lane & 15)]);
                b1[j] = bfrom(pp[k * DDd + nh + 16 + (lane & 15)]);
            }
            c0 = wmma_bf16(a, b0, c0);
            c1 = wmma_bf16(a, b1, c1);
        }
#pragma unroll
        for (int j = 0; j < 8; ++j) {
            int r = mt * 16 + j + ((lane >> 4) << 3);
            t1[r][nh + (lane & 15)]      = c0[j];
            t1[r][nh + 16 + (lane & 15)] = c1[j];
        }
    }
    __syncthreads();
    // ---- LN over 64 + relu -> t1b (pad rows come out as zeros) ----
    for (int r = wave; r < 64; r += 8) {
        float v0 = t1[r][2 * lane], v1 = t1[r][2 * lane + 1];
        float mean = wsum(v0 + v1) * (1.f / DDd);
        float d0 = v0 - mean, d1 = v1 - mean;
        float inv = rsqrtf(wsum(d0 * d0 + d1 * d1) * (1.f / DDd) + 1e-5f);
        t1b[r][2 * lane]     = f2bf(fmaxf(d0 * inv, 0.f));
        t1b[r][2 * lane + 1] = f2bf(fmaxf(d1 * inv, 0.f));
    }
    __syncthreads();
    // ---- phase 2: T2[49(pad64),256] = T1b[49,64] @ p2[64,256] ----
    {
        int mt = wave >> 1;
        int nbase = (wave & 1) * 128;
        for (int nt = 0; nt < 8; ++nt) {
            v8f c = {};
            int dcol = nbase + nt * 16 + (lane & 15);
            for (int kk = 0; kk < DDd; kk += 32) {
                int srow = mt * 16 + (lane & 15);
                int kb = (lane >> 4) << 4;
                v16bf a;
#pragma unroll
                for (int j = 0; j < 16; ++j) {
                    int k = kk + kb + j;
                    a[j] = bfrom(t1b[srow][k]);
                }
                int kbb = (lane >> 4) << 4;
                v16bf bb;
#pragma unroll
                for (int j = 0; j < 16; ++j) {
                    int e = kk + kbb + j;
                    bb[j] = bfrom(pp[Dd * DDd + e * Dd + dcol]);
                }
                c = wmma_bf16(a, bb, c);
            }
#pragma unroll
            for (int j = 0; j < 8; ++j)
                t2[mt * 16 + j + ((lane >> 4) << 3)][dcol] = c[j];
        }
    }
    __syncthreads();
    // ---- LN over 256 + relu -> global f2out ----
    for (int r = wave; r < Ss; r += 8) {
        float v[8];
        float loc = 0.f;
#pragma unroll
        for (int i = 0; i < 8; ++i) { v[i] = t2[r][lane + 32 * i]; loc += v[i]; }
        float mean = wsum(loc) * (1.f / Dd);
        float vv = 0.f;
#pragma unroll
        for (int i = 0; i < 8; ++i) { float d = v[i] - mean; vv += d * d; }
        float inv = rsqrtf(wsum(vv) * (1.f / Dd) + 1e-5f);
#pragma unroll
        for (int i = 0; i < 8; ++i) {
            float y = fmaxf((v[i] - mean) * inv, 0.f);
            f2out[(size_t)n * SD + r * Dd + lane + 32 * i] = f2bf(y);
        }
    }
}

// ---------------- host orchestration ----------------
typedef unsigned short ushort_t;

extern "C" void kernel_launch(void* const* d_in, const int* in_sizes, int n_in,
                              void* d_out, int out_size, void* d_ws, size_t ws_size,
                              hipStream_t stream) {
    const float* bboxes   = (const float*)d_in[0];
    const float* pro      = (const float*)d_in[1];
    const float* roi      = (const float*)d_in[2];
    const float* query    = (const float*)d_in[3];
    const float* cur      = (const float*)d_in[4];
    const float* w_qkv    = (const float*)d_in[5];
    const float* b_qkv    = (const float*)d_in[6];
    const float* w_attn   = (const float*)d_in[7];
    const float* b_attn   = (const float*)d_in[8];
    const float* w_dyn    = (const float*)d_in[9];
    const float* b_dyn    = (const float*)d_in[10];
    const float* w_dynout = (const float*)d_in[11];
    const float* b_dynout = (const float*)d_in[12];
    const float* w_ff1    = (const float*)d_in[13];
    const float* b_ff1    = (const float*)d_in[14];
    const float* w_ff2    = (const float*)d_in[15];
    const float* b_ff2    = (const float*)d_in[16];
    const float* w_cls    = (const float*)d_in[17];
    const float* w_logits = (const float*)d_in[18];
    const float* b_logits = (const float*)d_in[19];

    char* ws = (char*)d_ws;
    size_t off = 0;
    auto alloc = [&](size_t bytes) -> char* {
        char* p = ws + off;
        off += (bytes + 255) & ~(size_t)255;
        return p;
    };
    // bf16 weights
    ushort_t* wb_qkv    = (ushort_t*)alloc((size_t)768 * Dd * 2);
    ushort_t* wb_attn   = (ushort_t*)alloc((size_t)Dd * Dd * 2);
    ushort_t* wb_dyn    = (ushort_t*)alloc((size_t)PAR * Dd * 2);
    ushort_t* wb_dynout = (ushort_t*)alloc((size_t)Dd * SD * 2);
    ushort_t* wb_ff1    = (ushort_t*)alloc((size_t)FFf * Dd * 2);
    ushort_t* wb_ff2    = (ushort_t*)alloc((size_t)Dd * FFf * 2);
    ushort_t* wb_cls    = (ushort_t*)alloc((size_t)Dd * Dd * 2);
    ushort_t* wb_log    = (ushort_t*)alloc((size_t)NCc * Dd * 2);
    ushort_t* roi_bf    = (ushort_t*)alloc((size_t)Ss * NT * Dd * 2);
    // activations
    ushort_t* qk_bf   = (ushort_t*)alloc((size_t)NT * Dd * 2);
    ushort_t* pf_bf   = (ushort_t*)alloc((size_t)NT * Dd * 2);
    ushort_t* qkv_bf  = (ushort_t*)alloc((size_t)NT * 768 * 2);
    float*    maskb   = (float*)alloc((size_t)Nn * NR * NR * 4);
    ushort_t* ctx_bf  = (ushort_t*)alloc((size_t)NT * Dd * 2);
    float*    tgt2    = (float*)alloc((size_t)NT * Dd * 4);
    float*    pf1_f   = (float*)alloc((size_t)NT * Dd * 4);
    ushort_t* pf1_bf  = (ushort_t*)alloc((size_t)NT * Dd * 2);
    ushort_t* par_bf  = (ushort_t*)alloc((size_t)NT * PAR * 2);
    ushort_t* f2_bf   = (ushort_t*)alloc((size_t)NT * SD * 2);
    float*    gbuf    = (float*)alloc((size_t)NT * Dd * 4);
    float*    pfn2_f  = (float*)alloc((size_t)NT * Dd * 4);
    ushort_t* pfn2_bf = (ushort_t*)alloc((size_t)NT * Dd * 2);
    ushort_t* h_bf    = (ushort_t*)alloc((size_t)NT * FFf * 2);
    float*    t2buf   = (float*)alloc((size_t)NT * Dd * 4);
    ushort_t* fc_bf   = (ushort_t*)alloc((size_t)NT * Dd * 2);
    float*    clsg    = (float*)alloc((size_t)NT * Dd * 4);
    ushort_t* clsf_bf = (ushort_t*)alloc((size_t)NT * Dd * 2);

    auto cvt = [&](const float* s, ushort_t* d, long long n) {
        int grid = (int)((n + 255) / 256);
        if (grid > 4096) grid = 4096;
        f32_to_bf16_kernel<<<grid, 256, 0, stream>>>(s, d, n);
    };
    auto gemm = [&](const ushort_t* A, int lda, const ushort_t* W, int ldw,
                    const float* bias, float* oF, ushort_t* oB, int ldc,
                    int M, int N, int K, int relu) {
        dim3 g((M + 63) / 64, (N + 63) / 64);
        gemm_bf16_kernel<<<g, 256, 0, stream>>>(A, lda, W, ldw, bias, oF, oB, ldc, M, N, K, relu);
    };

    // weight + roi conversions
    cvt(w_qkv,    wb_qkv,    (long long)768 * Dd);
    cvt(w_attn,   wb_attn,   (long long)Dd * Dd);
    cvt(w_dyn,    wb_dyn,    (long long)PAR * Dd);
    cvt(w_dynout, wb_dynout, (long long)Dd * SD);
    cvt(w_ff1,    wb_ff1,    (long long)FFf * Dd);
    cvt(w_ff2,    wb_ff2,    (long long)Dd * FFf);
    cvt(w_cls,    wb_cls,    (long long)Dd * Dd);
    cvt(w_logits, wb_log,    (long long)NCc * Dd);
    cvt(roi,      roi_bf,    (long long)Ss * NT * Dd);

    // qk / pf prep + IoU mask bias
    qkprep_kernel<<<2048, 256, 0, stream>>>(pro, query, qk_bf, pf_bf, (long long)NT * Dd);
    mask_bias_kernel<<<Nn * NR, 128, 0, stream>>>(bboxes, cur, maskb);

    // q,k projections (cols 0..511) and v projection (cols 512..767), packed [NT,768]
    gemm(qk_bf, Dd, wb_qkv, Dd, b_qkv, nullptr, qkv_bf, 768, NT, 2 * Dd, Dd, 0);
    gemm(pf_bf, Dd, wb_qkv + (size_t)2 * Dd * Dd, Dd, b_qkv + 2 * Dd,
         nullptr, qkv_bf + 2 * Dd, 768, NT, Dd, Dd, 0);

    // attention
    attn_kernel<<<Nn * Hh * 19, 32, 0, stream>>>(qkv_bf, maskb, ctx_bf);

    // out-proj + norm1 (+ row mask)
    gemm(ctx_bf, Dd, wb_attn, Dd, b_attn, tgt2, nullptr, Dd, NT, Dd, Dd, 0);
    ln_rows_kernel<<<(NT + 7) / 8, 256, 0, stream>>>(tgt2, pro, cur, pf1_f, pf1_bf, NT, 0);

    // big parameter GEMM: [2400,256] @ [256,32768] -> bf16
    gemm(pf1_bf, Dd, wb_dyn, Dd, b_dyn, nullptr, par_bf, PAR, NT, PAR, Dd, 0);

    // fused DynamicConv (einsum1 + LN/relu + einsum2 + LN/relu) -> f2_bf
    dyn_interact_kernel<<<NT, 256, 16384 + 8192 + 65536, stream>>>(roi_bf, par_bf, f2_bf);

    // pf2 GEMM: [2400,12544] @ [12544,256] ; then norm2 chain
    gemm(f2_bf, SD, wb_dynout, SD, b_dynout, gbuf, nullptr, Dd, NT, Dd, SD, 0);
    ln_relu_ln_kernel<<<(NT + 7) / 8, 256, 0, stream>>>(gbuf, pf1_f, pfn2_f, pfn2_bf, NT);

    // FFN
    gemm(pfn2_bf, Dd, wb_ff1, Dd, b_ff1, nullptr, h_bf, FFf, NT, FFf, Dd, 1);
    gemm(h_bf, FFf, wb_ff2, FFf, b_ff2, t2buf, nullptr, Dd, NT, Dd, FFf, 0);
    // norm3 + mask -> fc
    ln_rows_kernel<<<(NT + 7) / 8, 256, 0, stream>>>(t2buf, pfn2_f, cur, nullptr, fc_bf, NT, 0);

    // cls tower: relu(ln(fc @ w_cls^T))
    gemm(fc_bf, Dd, wb_cls, Dd, nullptr, clsg, nullptr, Dd, NT, Dd, Dd, 0);
    ln_rows_kernel<<<(NT + 7) / 8, 256, 0, stream>>>(clsg, nullptr, nullptr, nullptr, clsf_bf, NT, 1);

    // logits -> d_out (f32 [2400,80])
    gemm(clsf_bf, Dd, wb_log, Dd, b_logits, (float*)d_out, nullptr, NCc, NT, NCc, Dd, 0);
}
